// Gmt_65609920413942
// MI455X (gfx1250) — matrix operand order repaired and checked by reading, not compiled, not run on hardware
//
#include <hip/hip_runtime.h>
#include <hip/hip_bf16.h>
#include <math.h>

// ---------------------------------------------------------------------------
// MI455X (gfx1250) implementation of the ViT block.
// GEMMs: V_WMMA_F32_16X16X32_BF16 with TDM (tensor_load_to_lds) double-buffered
// LDS staging of the activation tile. bf16 inputs, f32 accumulate.
// ---------------------------------------------------------------------------

typedef __bf16 bf16;
typedef __attribute__((ext_vector_type(16))) __bf16 v16bf;
typedef __attribute__((ext_vector_type(8)))  __bf16 v8bf;
typedef __attribute__((ext_vector_type(8)))  float  v8f;
typedef unsigned int v4u_t __attribute__((ext_vector_type(4)));
typedef int          v8i_t __attribute__((ext_vector_type(8)));
typedef int          v4i_t __attribute__((ext_vector_type(4)));

#define DEV static __device__ __forceinline__

// Problem constants
#define Bn    32
#define Cc    768
#define Hh    12
#define Dd    64
#define NPt   196
#define Mm    64
#define NGt   197          // 1 + NP
#define NT    261          // NG + M tokens per batch
#define ROWS  (Bn * NT)    // 8352 total rows, divisible by 32
#define VT_LD 264          // padded token dim for transposed V (16B-aligned v8bf)
#define GRIDQ 14
#define KSEG  256          // K-segment staged in LDS per TDM transfer

#if defined(__has_builtin)
#if __has_builtin(__builtin_amdgcn_tensor_load_to_lds) && \
    __has_builtin(__builtin_amdgcn_s_wait_tensorcnt)
#define HAVE_TDM 1
#endif
#endif
#if defined(__has_include)
#if __has_include(<hip/amd_detail/amd_gfx1250_TDM.h>)
#define TDM_6ARG 1
#endif
#endif

DEV v8bf ld8(const bf16* p) { return *(const v8bf*)p; }

DEV v16bf join16(v8bf lo, v8bf hi) {
  return __builtin_shufflevector(lo, hi, 0,1,2,3,4,5,6,7,8,9,10,11,12,13,14,15);
}

// A fragment (16 x 32, M x K), row-major source with leading dim ldk.
// lanes 0-15: row = r0+lane,    elems = K {k0..k0+7, k0+16..k0+23}
// lanes 16-31: row = r0+lane-16, elems = K {k0+8..k0+15, k0+24..k0+31}
DEV v16bf load_a(const bf16* base, int ldk, int r0, int k0, int lane) {
  int row = r0 + (lane & 15);
  int kb  = k0 + ((lane >> 4) << 3);      // +0 or +8
  const bf16* p = base + (size_t)row * ldk + kb;
  return join16(ld8(p), ld8(p + 16));
}

// B fragment (32 x 16, K x N) taken from row-major W[n][k] (i.e. B = W^T).
// lanes 0-15: n = n0+lane, K {k0..k0+7, k0+16..k0+23}; lanes 16-31: n = n0+lane-16, K {k0+8..k0+15, k0+24..k0+31}
DEV v16bf load_bT(const bf16* w, int ldk, int n0, int k0, int lane) {
  int n  = n0 + (lane & 15);
  int kb = k0 + ((lane >> 4) << 3);       // +0 or +8
  const bf16* p = w + (size_t)n * ldk + kb;
  return join16(ld8(p), ld8(p + 16));
}

DEV v8f wmma_bf16(v16bf a, v16bf b, v8f c) {
  return __builtin_amdgcn_wmma_f32_16x16x32_bf16(false, a, false, b, (short)0, c,
                                                 false, false);
}

DEV float gelu_exact(float v) {
  return 0.5f * v * (1.0f + erff(v * 0.70710678118654752f));
}

#ifdef HAVE_TDM
// Issue one TDM transfer: 32 rows x KSEG cols of bf16 from row-major A
// (leading dim Kelems) into LDS at byte offset lds_off.
// D# per CDNA5 ISA: 2D tile, data_size=2B.
DEV void tdm_load_a32xKSEG(const bf16* gsrc, unsigned lds_off, int Kelems) {
  unsigned long long ga = (unsigned long long)(uintptr_t)gsrc;
  v4u_t g0;
  g0[0] = 1u;                                   // count=1, is_restore=0, gather off
  g0[1] = lds_off;                              // lds_addr (bytes)
  g0[2] = (unsigned)ga;                         // global_addr[31:0]
  g0[3] = (unsigned)((ga >> 32) & 0x1FFFFFFu)   // global_addr[56:32]
          | (2u << 30);                         // type=2 ("image")
  v8i_t g1;
  g1[0] = 0x00010000;                           // wg_mask=0, data_size=1 (2B)
  g1[1] = (int)(((unsigned)Kelems & 0xFFFFu) << 16);        // tensor_dim0[15:0]
  g1[2] = (int)(((unsigned)Kelems >> 16) & 0xFFFFu)         // tensor_dim0[31:16]
          | (int)(((unsigned)ROWS & 0xFFFFu) << 16);        // tensor_dim1[15:0]
  g1[3] = (int)(((unsigned)ROWS >> 16) & 0xFFFFu)           // tensor_dim1[31:16]
          | (int)((unsigned)KSEG << 16);                    // tile_dim0 = KSEG
  g1[4] = 32;                                   // tile_dim1 = 32 rows, tile_dim2=0
  g1[5] = Kelems;                               // tensor_dim0_stride[31:0]
  g1[6] = 0;                                    // stride[47:32] | dim1_stride[15:0]
  g1[7] = 0;
  v4i_t z4 = {0, 0, 0, 0};
#ifdef TDM_6ARG
  v8i_t z8 = {0, 0, 0, 0, 0, 0, 0, 0};
  __builtin_amdgcn_tensor_load_to_lds(g0, g1, z4, z4, z8, 0);
#else
  __builtin_amdgcn_tensor_load_to_lds(g0, g1, z4, z4, 0);
#endif
}
#endif

// ---------------------------------------------------------------------------
// f32 -> bf16 conversion
// ---------------------------------------------------------------------------
__global__ void cvt_bf16_kernel(const float* __restrict__ src,
                                bf16* __restrict__ dst, int n) {
  int i = blockIdx.x * blockDim.x + threadIdx.x;
  if (i < n) dst[i] = (bf16)src[i];
}

// ---------------------------------------------------------------------------
// LayerNorm: one 256-thread block per row (768 = 3 * 256), bf16 output.
// ---------------------------------------------------------------------------
__global__ void ln_kernel(const float* __restrict__ x,
                          const float* __restrict__ w,
                          const float* __restrict__ b,
                          bf16* __restrict__ out) {
  int row = blockIdx.x;
  int t   = threadIdx.x;
  const float* xr = x + (size_t)row * Cc;
  float v0 = xr[t], v1 = xr[t + 256], v2 = xr[t + 512];

  __shared__ float red[256];
  red[t] = v0 + v1 + v2;
  __syncthreads();
  for (int o = 128; o > 0; o >>= 1) {
    if (t < o) red[t] += red[t + o];
    __syncthreads();
  }
  float mu = red[0] * (1.0f / Cc);
  __syncthreads();

  float d0 = v0 - mu, d1 = v1 - mu, d2 = v2 - mu;
  red[t] = d0 * d0 + d1 * d1 + d2 * d2;
  __syncthreads();
  for (int o = 128; o > 0; o >>= 1) {
    if (t < o) red[t] += red[t + o];
    __syncthreads();
  }
  float rs = rsqrtf(red[0] * (1.0f / Cc) + 1e-5f);

  bf16* orow = out + (size_t)row * Cc;
  orow[t]       = (bf16)(d0 * rs * w[t]       + b[t]);
  orow[t + 256] = (bf16)(d1 * rs * w[t + 256] + b[t + 256]);
  orow[t + 512] = (bf16)(d2 * rs * w[t + 512] + b[t + 512]);
}

// ---------------------------------------------------------------------------
// WMMA GEMM: out[ROWS x N] = A[ROWS x K] * W[N x K]^T  (+ epilogue)
//   MODE 0: scatter into q[b,h,n,d] / k[b,h,n,d] / vT[b,h,d,n] (bf16)
//   MODE 1: outf = resid + bias + acc            (proj + residual, f32)
//   MODE 2: outb = gelu(acc + bias)              (fc1, bf16)
//   MODE 3: outf += acc + bias                   (fc2 + residual, f32)
// Block: 256 threads = 8 waves; wave tile 16x64; block tile 32x256.
// A tile double-buffered in LDS via TDM (wave 0 drives the DMA engine);
// B (weights) streamed directly from global (L2-resident).
// ---------------------------------------------------------------------------
template <int MODE>
__global__ __launch_bounds__(256)
void gemm_kernel(const bf16* __restrict__ A,
                 const bf16* __restrict__ W, int K,
                 const float* __restrict__ bias,
                 const float* __restrict__ resid,
                 float* __restrict__ outf,
                 bf16* __restrict__ outb, int ldo,
                 bf16* __restrict__ qb,
                 bf16* __restrict__ kb,
                 bf16* __restrict__ vtb) {
  __shared__ bf16 As[2][32 * KSEG];   // 2 x 16 KB double buffer

  int lane = threadIdx.x & 31;
  int wid  = threadIdx.x >> 5;
  int rowg = wid >> 2;                // 0..1
  int colg = wid & 3;                 // 0..3
  int row0b = blockIdx.y * 32;
  int row0  = row0b + rowg * 16;
  int col0  = blockIdx.x * 256 + colg * 64;
  int rloc  = rowg * 16 + (lane & 15);       // A-frag row within LDS tile
  int ka_lo = (lane >> 4) << 3;              // A-frag K sub-offset (0 / 8)
  int nseg  = K / KSEG;

  v8f acc0 = {}, acc1 = {}, acc2 = {}, acc3 = {};

#ifdef HAVE_TDM
  unsigned lds_off[2] = {(unsigned)(uintptr_t)&As[0][0],
                         (unsigned)(uintptr_t)&As[1][0]};
  if (threadIdx.x < 32)
    tdm_load_a32xKSEG(A + (size_t)row0b * K, lds_off[0], K);
  for (int s = 0; s < nseg; ++s) {
    if (threadIdx.x < 32) {
      if (s + 1 < nseg) {
        tdm_load_a32xKSEG(A + (size_t)row0b * K + (size_t)(s + 1) * KSEG,
                          lds_off[(s + 1) & 1], K);
        __builtin_amdgcn_s_wait_tensorcnt(1);   // buffer s&1 landed
      } else {
        __builtin_amdgcn_s_wait_tensorcnt(0);
      }
    }
    __syncthreads();
    if (s + 1 < nseg)   // warm L2/WGP$ for next segment's weights
      __builtin_prefetch(W + (size_t)(col0 + (lane & 15)) * K + (s + 1) * KSEG,
                         0, 3);
    const bf16* Abuf = As[s & 1];
    for (int kk = 0; kk < KSEG; kk += 32) {
      int k0 = s * KSEG + kk;
      const bf16* ap = Abuf + (size_t)rloc * KSEG + kk + ka_lo;
      v16bf a  = join16(ld8(ap), ld8(ap + 16));     // ds_load_b128 x2
      v16bf b0 = load_bT(W, K, col0,      k0, lane);
      v16bf b1 = load_bT(W, K, col0 + 16, k0, lane);
      v16bf b2 = load_bT(W, K, col0 + 32, k0, lane);
      v16bf b3 = load_bT(W, K, col0 + 48, k0, lane);
      acc0 = wmma_bf16(a, b0, acc0);
      acc1 = wmma_bf16(a, b1, acc1);
      acc2 = wmma_bf16(a, b2, acc2);
      acc3 = wmma_bf16(a, b3, acc3);
    }
    __syncthreads();    // readers done before buffer s&1 is re-DMA'd
  }
#else
  // Fallback: cooperative global->LDS staging (single buffer).
  for (int s = 0; s < nseg; ++s) {
    __syncthreads();
    for (int c = threadIdx.x; c < 32 * KSEG / 8; c += 256) {
      int r = (c * 8) / KSEG, cc = (c * 8) % KSEG;
      *(v8bf*)&As[0][r * KSEG + cc] =
          ld8(A + (size_t)(row0b + r) * K + s * KSEG + cc);
    }
    __syncthreads();
    for (int kk = 0; kk < KSEG; kk += 32) {
      int k0 = s * KSEG + kk;
      const bf16* ap = As[0] + (size_t)rloc * KSEG + kk + ka_lo;
      v16bf a  = join16(ld8(ap), ld8(ap + 16));
      v16bf b0 = load_bT(W, K, col0,      k0, lane);
      v16bf b1 = load_bT(W, K, col0 + 16, k0, lane);
      v16bf b2 = load_bT(W, K, col0 + 32, k0, lane);
      v16bf b3 = load_bT(W, K, col0 + 48, k0, lane);
      acc0 = wmma_bf16(a, b0, acc0);
      acc1 = wmma_bf16(a, b1, acc1);
      acc2 = wmma_bf16(a, b2, acc2);
      acc3 = wmma_bf16(a, b3, acc3);
    }
  }
#endif

  int hi8 = (lane >> 4) << 3;   // 0 or 8 (row offset inside C tile)
  int nlo = lane & 15;
  v8f accs[4] = {acc0, acc1, acc2, acc3};
#pragma unroll
  for (int j = 0; j < 4; ++j) {
#pragma unroll
    for (int r = 0; r < 8; ++r) {
      int row = row0 + r + hi8;
      int col = col0 + j * 16 + nlo;
      float val = accs[j][r];
      if (MODE == 0) {
        int which = col / Cc;
        int c     = col % Cc;
        int head  = c >> 6;
        int d     = c & 63;
        int bb    = row / NT;
        int n     = row % NT;
        size_t bh = (size_t)(bb * Hh + head);
        if (which == 0)       qb[(bh * NT + n) * Dd + d]     = (bf16)val;
        else if (which == 1)  kb[(bh * NT + n) * Dd + d]     = (bf16)val;
        else                  vtb[(bh * Dd + d) * VT_LD + n] = (bf16)val;
      } else if (MODE == 1) {
        size_t o = (size_t)row * ldo + col;
        outf[o] = resid[o] + bias[col] + val;
      } else if (MODE == 2) {
        outb[(size_t)row * ldo + col] = (bf16)gelu_exact(val + bias[col]);
      } else {
        size_t o = (size_t)row * ldo + col;
        outf[o] += val + bias[col];
      }
    }
  }
}

// ---------------------------------------------------------------------------
// Global attention over the first 197 tokens. One wave (32 threads) per
// (b, h, query-tile-of-16). S = q k^T * 0.125 via WMMA, softmax staged in
// LDS (also performs the C-layout -> A-layout transpose), then P @ V via
// WMMA against transposed V.
// ---------------------------------------------------------------------------
__global__ void attn_global_kernel(const bf16* __restrict__ q,
                                   const bf16* __restrict__ k,
                                   const bf16* __restrict__ vT,
                                   bf16* __restrict__ attn_out) {
  int lane = threadIdx.x & 31;
  int bid  = blockIdx.x;
  int t    = bid % 13;            // query tile (13 tiles cover 0..207)
  int h    = (bid / 13) % Hh;
  int b    = bid / (13 * Hh);
  int q0   = t * 16;

  const bf16* qB = q  + ((size_t)(b * Hh + h)) * NT * Dd;
  const bf16* kB = k  + ((size_t)(b * Hh + h)) * NT * Dd;
  const bf16* vB = vT + ((size_t)(b * Hh + h)) * Dd * VT_LD;

  __shared__ float Ssm[16][208];
  __shared__ bf16  Pls[16][224];

  v16bf aq0 = load_a(qB, Dd, q0, 0,  lane);
  v16bf aq1 = load_a(qB, Dd, q0, 32, lane);

  int hi8 = (lane >> 4) << 3;
  int nlo = lane & 15;

  for (int kt = 0; kt < 13; ++kt) {
    v16bf bk0 = load_bT(kB, Dd, kt * 16, 0,  lane);
    v16bf bk1 = load_bT(kB, Dd, kt * 16, 32, lane);
    v8f s = {};
    s = wmma_bf16(aq0, bk0, s);
    s = wmma_bf16(aq1, bk1, s);
    int key = kt * 16 + nlo;
#pragma unroll
    for (int r = 0; r < 8; ++r) {
      int m = r + hi8;
      Ssm[m][key] = (key < NGt) ? s[r] * 0.125f : -1e30f;
    }
  }
  __syncthreads();

  if (lane < 16) {
    float mx = -1e30f;
    for (int j = 0; j < 208; ++j) mx = fmaxf(mx, Ssm[lane][j]);
    float sum = 0.0f;
    for (int j = 0; j < 208; ++j) sum += expf(Ssm[lane][j] - mx);
    float inv = 1.0f / sum;
    for (int j = 0; j < 208; ++j)
      Pls[lane][j] = (bf16)(expf(Ssm[lane][j] - mx) * inv);
    for (int j = 208; j < 224; ++j) Pls[lane][j] = (bf16)0.0f;
  }
  __syncthreads();

  v8f co0 = {}, co1 = {}, co2 = {}, co3 = {};
  for (int ks = 0; ks < 7; ++ks) {
    int kk = ks * 32;
    int m  = lane & 15;
    int ka = kk + ((lane >> 4) << 3);
    v16bf ap = join16(ld8(&Pls[m][ka]), ld8(&Pls[m][ka + 16]));
    int koff = kk + ((lane >> 4) << 3);
#pragma unroll
    for (int j = 0; j < 4; ++j) {
      int d = j * 16 + nlo;
      const bf16* p = vB + (size_t)d * VT_LD + koff;
      v16bf bv = join16(ld8(p), ld8(p + 16));
      if (j == 0) co0 = wmma_bf16(ap, bv, co0);
      else if (j == 1) co1 = wmma_bf16(ap, bv, co1);
      else if (j == 2) co2 = wmma_bf16(ap, bv, co2);
      else co3 = wmma_bf16(ap, bv, co3);
    }
  }

  v8f cos[4] = {co0, co1, co2, co3};
#pragma unroll
  for (int j = 0; j < 4; ++j) {
#pragma unroll
    for (int r = 0; r < 8; ++r) {
      int qq = q0 + r + hi8;
      if (qq < NGt) {
        int d = j * 16 + nlo;
        attn_out[((size_t)(b * NT + qq)) * Cc + h * Dd + d] = (bf16)cos[j][r];
      }
    }
  }
}

// ---------------------------------------------------------------------------
// Point attention: 4 bilinear-corner neighbors per point token.
// Block = 384 threads (12 waves = 12 heads); grid = B * M.
// ---------------------------------------------------------------------------
__global__ void attn_point_kernel(const float* __restrict__ ps,
                                  const bf16* __restrict__ q,
                                  const bf16* __restrict__ k,
                                  const bf16* __restrict__ vT,
                                  bf16* __restrict__ attn_out) {
  int lane = threadIdx.x & 31;
  int h    = threadIdx.x >> 5;
  int m    = blockIdx.x % Mm;
  int b    = blockIdx.x / Mm;

  float px = ps[((size_t)(b * Mm + m)) * 2 + 0] * (1.0f / 16.0f);
  float py = ps[((size_t)(b * Mm + m)) * 2 + 1] * (1.0f / 16.0f);
  int c0 = (int)fminf(fmaxf(floorf(px), 0.0f), (float)(GRIDQ - 1));
  int c1 = (int)fminf(fmaxf(ceilf(px),  0.0f), (float)(GRIDQ - 1));
  int r0 = (int)fminf(fmaxf(floorf(py), 0.0f), (float)(GRIDQ - 1));
  int r1 = (int)fminf(fmaxf(ceilf(py),  0.0f), (float)(GRIDQ - 1));
  int tok[4];
  tok[0] = 1 + r0 * GRIDQ + c0;
  tok[1] = 1 + r1 * GRIDQ + c0;
  tok[2] = 1 + r0 * GRIDQ + c1;
  tok[3] = 1 + r1 * GRIDQ + c1;

  size_t bh   = (size_t)(b * Hh + h);
  size_t qoff = (bh * NT + (NGt + m)) * Dd;
  float q0 = (float)q[qoff + lane];
  float q1 = (float)q[qoff + 32 + lane];

  float a[4];
#pragma unroll
  for (int j = 0; j < 4; ++j) {
    size_t ko = (bh * NT + tok[j]) * Dd;
    float dot = q0 * (float)k[ko + lane] + q1 * (float)k[ko + 32 + lane];
#pragma unroll
    for (int o = 16; o > 0; o >>= 1) dot += __shfl_xor(dot, o, 32);
    a[j] = dot;
  }
  float mx = fmaxf(fmaxf(a[0], a[1]), fmaxf(a[2], a[3]));
  float e0 = expf(a[0] - mx), e1 = expf(a[1] - mx);
  float e2 = expf(a[2] - mx), e3 = expf(a[3] - mx);
  float inv = 1.0f / (e0 + e1 + e2 + e3);
  float p[4] = {e0 * inv, e1 * inv, e2 * inv, e3 * inv};

  float o0 = 0.0f, o1 = 0.0f;
#pragma unroll
  for (int j = 0; j < 4; ++j) {
    o0 += p[j] * (float)vT[(bh * Dd + lane) * VT_LD + tok[j]];
    o1 += p[j] * (float)vT[(bh * Dd + 32 + lane) * VT_LD + tok[j]];
  }
  size_t orow = ((size_t)(b * NT + NGt + m)) * Cc + h * Dd;
  attn_out[orow + lane]      = (bf16)o0;
  attn_out[orow + 32 + lane] = (bf16)o1;
}

// ---------------------------------------------------------------------------
// Host-side orchestration
// ---------------------------------------------------------------------------
extern "C" void kernel_launch(void* const* d_in, const int* in_sizes, int n_in,
                              void* d_out, int out_size, void* d_ws, size_t ws_size,
                              hipStream_t stream) {
  const float* x       = (const float*)d_in[0];
  const float* ps      = (const float*)d_in[1];
  // d_in[2] = ns (unused by the reference math)
  const float* norm1_w = (const float*)d_in[3];
  const float* norm1_b = (const float*)d_in[4];
  const float* qkv_w   = (const float*)d_in[5];
  const float* proj_w  = (const float*)d_in[6];
  const float* proj_b  = (const float*)d_in[7];
  const float* norm2_w = (const float*)d_in[8];
  const float* norm2_b = (const float*)d_in[9];
  const float* fc1_w   = (const float*)d_in[10];
  const float* fc1_b   = (const float*)d_in[11];
  const float* fc2_w   = (const float*)d_in[12];
  const float* fc2_b   = (const float*)d_in[13];
  float* out = (float*)d_out;

  char* ws = (char*)d_ws;
  size_t off = 0;
  auto take = [&](size_t bytes) -> char* {
    char* p = ws + off;
    off += (bytes + 255) & ~(size_t)255;
    return p;
  };
  bf16* qkvw_b = (bf16*)take((size_t)3 * Cc * Cc * 2);          // 2304 x 768
  bf16* projw_b = (bf16*)take((size_t)Cc * Cc * 2);             // 768 x 768
  bf16* fc1w_b = (bf16*)take((size_t)4 * Cc * Cc * 2);          // 3072 x 768
  bf16* fc2w_b = (bf16*)take((size_t)Cc * 4 * Cc * 2);          // 768 x 3072
  bf16* h_b    = (bf16*)take((size_t)ROWS * Cc * 2);
  bf16* q_b    = (bf16*)take((size_t)Bn * Hh * NT * Dd * 2);
  bf16* k_b    = (bf16*)take((size_t)Bn * Hh * NT * Dd * 2);
  bf16* vT_b   = (bf16*)take((size_t)Bn * Hh * Dd * VT_LD * 2);
  bf16* ao_b   = (bf16*)take((size_t)ROWS * Cc * 2);            // concat(xg,kout)
  bf16* h2_b   = (bf16*)take((size_t)ROWS * Cc * 2);
  bf16* g_b    = (bf16*)take((size_t)ROWS * 4 * Cc * 2);        // fc1 output

  // 1) weights -> bf16
  {
    int n;
    n = 3 * Cc * Cc; cvt_bf16_kernel<<<(n + 255) / 256, 256, 0, stream>>>(qkv_w, qkvw_b, n);
    n = Cc * Cc;     cvt_bf16_kernel<<<(n + 255) / 256, 256, 0, stream>>>(proj_w, projw_b, n);
    n = 4 * Cc * Cc; cvt_bf16_kernel<<<(n + 255) / 256, 256, 0, stream>>>(fc1_w, fc1w_b, n);
    n = 4 * Cc * Cc; cvt_bf16_kernel<<<(n + 255) / 256, 256, 0, stream>>>(fc2_w, fc2w_b, n);
  }

  // 2) LN1
  ln_kernel<<<ROWS, 256, 0, stream>>>(x, norm1_w, norm1_b, h_b);

  // 3) QKV GEMM (8352 x 768) * (2304 x 768)^T, scatter epilogue
  gemm_kernel<0><<<dim3(3 * Cc / 256, ROWS / 32), 256, 0, stream>>>(
      h_b, qkvw_b, Cc, nullptr, nullptr, nullptr, nullptr, 0, q_b, k_b, vT_b);

  // 4) global attention (WMMA flash-style, one wave per 16-query tile)
  attn_global_kernel<<<Bn * Hh * 13, 32, 0, stream>>>(q_b, k_b, vT_b, ao_b);

  // 5) point attention
  attn_point_kernel<<<Bn * Mm, Hh * 32, 0, stream>>>(ps, q_b, k_b, vT_b, ao_b);

  // 6) proj GEMM + bias + residual -> d_out (f32)
  gemm_kernel<1><<<dim3(Cc / 256, ROWS / 32), 256, 0, stream>>>(
      ao_b, projw_b, Cc, proj_b, x, out, nullptr, Cc, nullptr, nullptr, nullptr);

  // 7) LN2 on x1 = d_out
  ln_kernel<<<ROWS, 256, 0, stream>>>(out, norm2_w, norm2_b, h2_b);

  // 8) FC1 + exact GELU -> bf16
  gemm_kernel<2><<<dim3(4 * Cc / 256, ROWS / 32), 256, 0, stream>>>(
      h2_b, fc1w_b, Cc, fc1_b, nullptr, nullptr, g_b, 4 * Cc, nullptr, nullptr, nullptr);

  // 9) FC2 + bias, accumulate into d_out
  gemm_kernel<3><<<dim3(Cc / 256, ROWS / 32), 256, 0, stream>>>(
      g_b, fc2w_b, 4 * Cc, fc2_b, nullptr, out, nullptr, Cc, nullptr, nullptr, nullptr);

  (void)in_sizes; (void)n_in; (void)out_size; (void)ws_size;
}
